// EdgeBlock_33071248179443
// MI455X (gfx1250) — matrix operand order repaired and checked by reading, not compiled, not run on hardware
//
#include <hip/hip_runtime.h>

// ---------------------------------------------------------------------------
// EdgeBlock: out = relu(concat(S,R,E) @ W1 + b1) @ W2 + b2
// Transposed-operand formulation: D' = W^T (A operand) x X^T (B operand),
// f16 WMMA (V_WMMA_F32_16X16X32_F16), f32 accumulate. Wave32 / gfx1250.
// 4 waves/WG, each wave: 2 M-tiles (32 edges) x 8 N-tiles (128 features).
// Weight A-fragments feed 2 WMMAs; activation B-fragments feed 8 WMMAs.
// C layout now gives each lane 8 consecutive output features of one edge:
// b128 LDS/global epilogues, biases folded into accumulator init.
// ---------------------------------------------------------------------------

typedef __attribute__((ext_vector_type(16))) _Float16 v16h;
typedef __attribute__((ext_vector_type(8)))  _Float16 v8h;
typedef __attribute__((ext_vector_type(8)))  float    v8f;
typedef __attribute__((ext_vector_type(4)))  float    v4f;

#define DD      128      // feature dim
#define TILE_E  128      // edges per workgroup tile (4 waves x 32 edges)
#define NTILES  8        // 128/16 feature tiles per wave
#define MT      2        // edge (M) tiles per wave
#define NBLK    1250     // persistent grid

// B fragment (32x16 KxN f16): lane = one activation row (edge), 16 contiguous
// K values starting at kbase + (lane>=16 ? 16 : 0). p points at those 16 floats.
template <bool NT>
__device__ __forceinline__ v16h cvt_b_frag(const float* p) {
  v4f x0, x1, x2, x3;
  if (NT) {
    x0 = __builtin_nontemporal_load((const v4f*)(p));
    x1 = __builtin_nontemporal_load((const v4f*)(p + 4));
    x2 = __builtin_nontemporal_load((const v4f*)(p + 8));
    x3 = __builtin_nontemporal_load((const v4f*)(p + 12));
  } else {
    x0 = *(const v4f*)(p);
    x1 = *(const v4f*)(p + 4);
    x2 = *(const v4f*)(p + 8);
    x3 = *(const v4f*)(p + 12);
  }
  v16h r;
#pragma unroll
  for (int i = 0; i < 4; ++i) {
    r[i]      = (_Float16)x0[i];
    r[4 + i]  = (_Float16)x1[i];
    r[8 + i]  = (_Float16)x2[i];
    r[12 + i] = (_Float16)x3[i];
  }
  return r;
}

// 16 halves via two b128 LDS loads.
__device__ __forceinline__ v16h load_h16(const _Float16* p0, const _Float16* p1) {
  union { v16h v; v8h h[2]; } u;
  u.h[0] = *(const v8h*)p0;
  u.h[1] = *(const v8h*)p1;
  return u.v;
}

// Per-lane v8f of 8 consecutive floats.
__device__ __forceinline__ v8f load8f(const float* p) {
  v4f l0 = *(const v4f*)p;
  v4f l1 = *(const v4f*)(p + 4);
  return __builtin_shufflevector(l0, l1, 0, 1, 2, 3, 4, 5, 6, 7);
}

__global__ __launch_bounds__(128)
void edgeblock_wmma_kernel(const float* __restrict__ node_attr,
                           const int*   __restrict__ edge_index,
                           const float* __restrict__ edge_attr,
                           const float* __restrict__ w1,
                           const float* __restrict__ b1,
                           const float* __restrict__ w2,
                           const float* __restrict__ b2,
                           float*       __restrict__ out,
                           int n_edges, int n_nodes, int n_tiles)
{
  // LDS: W1 (384x128 f16, swizzled) + W2 (128x128 f16, swizzled) + h (128x128 f16)
  // Swizzle: [kc][n][k%32] -> K contiguous per output-feature n => weight
  // A-fragments are ds_load_b128 pairs. 160 KB total -> 2 WGs per 320 KB WGP.
  __shared__ _Float16 w1s[12 * 128 * 32];        // 96 KB
  __shared__ _Float16 w2s[ 4 * 128 * 32];        // 32 KB
  __shared__ _Float16 h_lds[TILE_E * DD];        // 32 KB

  const int tid = threadIdx.x;

  // ---- Stage + convert + swizzle weights once per workgroup ----
  for (int g = tid; g < (384 / 4) * 128; g += 128) {
    int n  = g & 127;
    int kg = g >> 7;            // group of 4 consecutive K
    int k  = kg * 4;
    float f0 = w1[(k + 0) * DD + n];
    float f1 = w1[(k + 1) * DD + n];
    float f2 = w1[(k + 2) * DD + n];
    float f3 = w1[(k + 3) * DD + n];
    _Float16* dst = &w1s[((k >> 5) * 128 + n) * 32 + (k & 31)];
    dst[0] = (_Float16)f0; dst[1] = (_Float16)f1;
    dst[2] = (_Float16)f2; dst[3] = (_Float16)f3;
  }
  for (int g = tid; g < (128 / 4) * 128; g += 128) {
    int n  = g & 127;
    int kg = g >> 7;
    int k  = kg * 4;
    float f0 = w2[(k + 0) * DD + n];
    float f1 = w2[(k + 1) * DD + n];
    float f2 = w2[(k + 2) * DD + n];
    float f3 = w2[(k + 3) * DD + n];
    _Float16* dst = &w2s[((k >> 5) * 128 + n) * 32 + (k & 31)];
    dst[0] = (_Float16)f0; dst[1] = (_Float16)f1;
    dst[2] = (_Float16)f2; dst[3] = (_Float16)f3;
  }
  __syncthreads();

  const int lane = tid & 31;
  const int wave = tid >> 5;    // 0..3
  const int mcol = lane & 15;   // edge row within 16-edge tile (C column / B column)
  const int half = lane >> 4;   // lane group (0: lanes 0-15, 1: lanes 16-31)

  const int* senders   = edge_index;
  const int* receivers = edge_index + n_edges;

  for (int t = blockIdx.x; t < n_tiles; t += gridDim.x) {
    const int edge0 = t * TILE_E + wave * (MT * 16);

    const float* rowS[MT];
    const float* rowR[MT];
    const float* rowE[MT];
#pragma unroll
    for (int mt = 0; mt < MT; ++mt) {
      const int my_edge = edge0 + mt * 16 + mcol;
      const int ce      = my_edge < n_edges ? my_edge : n_edges - 1;
      rowS[mt] = node_attr + (size_t)senders[ce]   * DD;
      rowR[mt] = node_attr + (size_t)receivers[ce] * DD;
      rowE[mt] = edge_attr + (size_t)ce            * DD;
    }

    // Fold b1 into accumulator init: acc element j of tile nt is output
    // feature n = nt*16 + half*8 + j (same for every lane / edge).
    v8f acc[MT][NTILES];
#pragma unroll
    for (int nt = 0; nt < NTILES; ++nt) {
      v8f bv = load8f(b1 + nt * 16 + half * 8);
#pragma unroll
      for (int mt = 0; mt < MT; ++mt) acc[mt][nt] = bv;
    }

    // ---- GEMM 1: h = relu(S@W1a + R@W1b + E@W1c + b1), transposed operands ----
#pragma unroll
    for (int s = 0; s < 3; ++s) {
#pragma unroll
      for (int kc = 0; kc < 4; ++kc) {
        v16h bf[MT];
#pragma unroll
        for (int mt = 0; mt < MT; ++mt) {
          const float* src = (s == 0) ? rowS[mt] : (s == 1) ? rowR[mt] : rowE[mt];
          const float* p = src + kc * 32 + half * 16;
          bf[mt] = (s == 2) ? cvt_b_frag<true>(p) : cvt_b_frag<false>(p);
        }
        const int kcb = s * 4 + kc;
#pragma unroll
        for (int nt = 0; nt < NTILES; ++nt) {
          const int n = nt * 16 + mcol;       // A' row = output feature
          const _Float16* ap = &w1s[(kcb * 128 + n) * 32];
          v16h a = load_h16(ap + half * 16, ap + half * 16 + 8);
#pragma unroll
          for (int mt = 0; mt < MT; ++mt) {
            acc[mt][nt] = __builtin_amdgcn_wmma_f32_16x16x32_f16(
                false, a, false, bf[mt], (short)0, acc[mt][nt], false, false);
          }
        }
      }
    }

    // ---- ReLU + pack, h row-major to per-wave LDS region (b128 stores) ----
#pragma unroll
    for (int mt = 0; mt < MT; ++mt) {
      _Float16* hb = &h_lds[((wave * MT + mt) * 16 + mcol) * DD];
#pragma unroll
      for (int nt = 0; nt < NTILES; ++nt) {
        v8h hh;
#pragma unroll
        for (int j = 0; j < 8; ++j) {
          float x = acc[mt][nt][j];
          x = x > 0.0f ? x : 0.0f;
          hh[j] = (_Float16)x;
        }
        *(v8h*)(hb + nt * 16 + half * 8) = hh;
      }
    }
    // Per-wave LDS RAW: wait for DS stores, no workgroup barrier needed.
    asm volatile("s_wait_dscnt 0x0" ::: "memory");

    // ---- GEMM 2: out = h @ W2 + b2, transposed operands ----
    v8f acc2[MT][NTILES];
#pragma unroll
    for (int nt = 0; nt < NTILES; ++nt) {
      v8f bv = load8f(b2 + nt * 16 + half * 8);
#pragma unroll
      for (int mt = 0; mt < MT; ++mt) acc2[mt][nt] = bv;
    }

#pragma unroll
    for (int kc = 0; kc < 4; ++kc) {
      v16h bf[MT];
#pragma unroll
      for (int mt = 0; mt < MT; ++mt) {
        const _Float16* hp =
            &h_lds[((wave * MT + mt) * 16 + mcol) * DD + kc * 32 + half * 16];
        bf[mt] = load_h16(hp, hp + 8);
      }
#pragma unroll
      for (int nt = 0; nt < NTILES; ++nt) {
        const int n = nt * 16 + mcol;
        const _Float16* ap = &w2s[(kc * 128 + n) * 32];
        v16h a = load_h16(ap + half * 16, ap + half * 16 + 8);
#pragma unroll
        for (int mt = 0; mt < MT; ++mt) {
          acc2[mt][nt] = __builtin_amdgcn_wmma_f32_16x16x32_f16(
              false, a, false, bf[mt], (short)0, acc2[mt][nt], false, false);
        }
      }
    }

    // ---- streaming (non-temporal) b128 f32 stores: 8 consecutive features ----
#pragma unroll
    for (int mt = 0; mt < MT; ++mt) {
      const int e = edge0 + mt * 16 + mcol;
      if (e < n_edges) {
        float* op = out + (size_t)e * DD + half * 8;
#pragma unroll
        for (int nt = 0; nt < NTILES; ++nt) {
          v4f lo = __builtin_shufflevector(acc2[mt][nt], acc2[mt][nt], 0, 1, 2, 3);
          v4f hi = __builtin_shufflevector(acc2[mt][nt], acc2[mt][nt], 4, 5, 6, 7);
          __builtin_nontemporal_store(lo, (v4f*)(op + nt * 16));
          __builtin_nontemporal_store(hi, (v4f*)(op + nt * 16 + 4));
        }
      }
    }
  }
}

extern "C" void kernel_launch(void* const* d_in, const int* in_sizes, int n_in,
                              void* d_out, int out_size, void* d_ws, size_t ws_size,
                              hipStream_t stream) {
  const float* node_attr  = (const float*)d_in[0];
  const int*   edge_index = (const int*)  d_in[1];
  const float* edge_attr  = (const float*)d_in[2];
  const float* w1         = (const float*)d_in[3];
  const float* b1         = (const float*)d_in[4];
  const float* w2         = (const float*)d_in[5];
  const float* b2         = (const float*)d_in[6];
  float*       out        = (float*)d_out;

  const int n_nodes = in_sizes[0] / DD;
  const int n_edges = in_sizes[1] / 2;
  const int n_tiles = (n_edges + TILE_E - 1) / TILE_E;
  const int nblk    = n_tiles < NBLK ? n_tiles : NBLK;

  edgeblock_wmma_kernel<<<dim3(nblk), dim3(128), 0, stream>>>(
      node_attr, edge_index, edge_attr, w1, b1, w2, b2, out,
      n_edges, n_nodes, n_tiles);
}
